// MyRnn2_53128745452127
// MI455X (gfx1250) — compile-verified, matrix operand order checked by reading, not run on hardware
//
#include <hip/hip_runtime.h>
#include <hip/hip_bf16.h>

typedef _Float16 v16h __attribute__((ext_vector_type(16)));
typedef _Float16 v8h  __attribute__((ext_vector_type(8)));
typedef float    v8f  __attribute__((ext_vector_type(8)));

#define HID  128
#define G3   384
#define BT   32      // batch tile per workgroup
#define SEQ  256
#define KIN  28
#define KPAD 32
#define OUTN 10

// f16 workspace layout (element offsets)
#define WS_WIH1 0                         // [384][32]  (K padded 28->32 with zeros)
#define WS_WHH1 12288                     // [384][128]
#define WS_WIH2 61440                     // [384][128]
#define WS_WHH2 110592                    // [384][128]
#define WS_TOTAL 159744

__global__ void prep_weights(const float* __restrict__ wih1,
                             const float* __restrict__ whh1,
                             const float* __restrict__ wih2,
                             const float* __restrict__ whh2,
                             _Float16* __restrict__ ws) {
    int i = blockIdx.x * 256 + threadIdx.x;
    if (i < WS_WHH1) {
        int r = i >> 5, k = i & 31;
        ws[i] = (k < KIN) ? (_Float16)wih1[r * KIN + k] : (_Float16)0.f;
    } else if (i < WS_WIH2) {
        ws[i] = (_Float16)whh1[i - WS_WHH1];
    } else if (i < WS_WHH2) {
        ws[i] = (_Float16)wih2[i - WS_WIH2];
    } else if (i < WS_TOTAL) {
        ws[i] = (_Float16)whh2[i - WS_WHH2];
    }
}

// Load a 16x32 f16 WMMA fragment from a row-major [rows][pitch] f16 array.
// Per ISA layout: lanes 0-15 hold row 'row', K chunks {kbase..+7, kbase+16..+23};
// lanes 16-31 hold chunks offset by +8.  Caller folds (lane&15) into 'row' and
// (lane>>4)*8 into 'kbase'.  B fragments use the same pattern on W stored [N][K].
__device__ __forceinline__ v16h load_frag(const _Float16* base, int pitch, int row, int kbase) {
    const _Float16* p = base + row * pitch + kbase;
    v8h lo = *(const v8h*)(p);
    v8h hi = *(const v8h*)(p + 16);
    return __builtin_shufflevector(lo, hi, 0,1,2,3,4,5,6,7,8,9,10,11,12,13,14,15);
}

// minimal-VALU activations: raw v_exp_f32 (exp2) + v_rcp_f32
#define LOG2E  1.4426950408889634f
#define LOG2E2 2.8853900817779268f
__device__ __forceinline__ float fsigm(float x) {   // 1/(1+e^-x): mul,exp,add,rcp
    return __builtin_amdgcn_rcpf(1.f + __builtin_amdgcn_exp2f(-x * LOG2E));
}
__device__ __forceinline__ float ftanh(float x) {   // 2*sigm(2x)-1: 5 VALU
    float t = __builtin_amdgcn_rcpf(1.f + __builtin_amdgcn_exp2f(-x * LOG2E2));
    return __builtin_fmaf(2.f, t, -1.f);
}

__device__ __forceinline__ v8f bcast8(float b) {
    v8f c;
#pragma unroll
    for (int e = 0; e < 8; ++e) c[e] = b;
    return c;
}

#define WMMA(A, B, C) __builtin_amdgcn_wmma_f32_16x16x32_f16(false, (A), false, (B), (short)0, (C), false, false)

__global__ void __launch_bounds__(256) gru_main(
        const float* __restrict__ x,          // [n,1,256,28] fp32
        const _Float16* __restrict__ ws,      // f16 weights (prep_weights layout)
        const float* __restrict__ bi1, const float* __restrict__ bh1,
        const float* __restrict__ bi2, const float* __restrict__ bh2,
        const float* __restrict__ wout,       // [10][128] fp32
        const float* __restrict__ bout,       // [10] fp32
        float* __restrict__ out)              // [n][10] fp32
{
    const int tid  = threadIdx.x;
    const int wave = tid >> 5;
    const int lane = tid & 31;
    const int half = lane >> 4;        // selects +8 K-chunk half of fragment
    const int lcol = lane & 15;        // row (A) / column (B,C,D) within 16-tile
    const int kb   = half * 8;
    const int b0   = blockIdx.x * BT;
    const int col  = wave * 16 + lcol; // this wave's hidden-column slice

    // ---- LDS carve (dynamic, 215040 B < 320 KB WGP LDS) ----
    extern __shared__ _Float16 smem[];
    _Float16* sW1 = smem;              // w_hh1 [384][128]  (49152)
    _Float16* sW2 = smem + 49152;      // w_hh2 [384][128]  (49152)
    _Float16* sH1 = smem + 98304;      // h1    [32][128]   (4096)
    _Float16* sH2 = smem + 102400;     // h2    [32][128]   (4096)
    _Float16* sX  = smem + 106496;     // x_t   [32][32]    (1024)

    // stage recurrent weights to LDS, zero states + x pad
    {
        const uint4* s1 = (const uint4*)(ws + WS_WHH1);
        const uint4* s2 = (const uint4*)(ws + WS_WHH2);
        uint4* d1 = (uint4*)sW1;
        uint4* d2 = (uint4*)sW2;
        for (int i = tid; i < 6144; i += 256) d1[i] = s1[i];
        for (int i = tid; i < 6144; i += 256) d2[i] = s2[i];
        uint4* z = (uint4*)sH1;   // sH1, sH2, sX contiguous: 9216 f16 = 1152 uint4
        for (int i = tid; i < 1152; i += 256) z[i] = make_uint4(0, 0, 0, 0);
    }

    // register-resident B fragments: w_ih1 (K=32) and w_ih2 (K=128)
    v16h fI1[3];
#pragma unroll
    for (int g = 0; g < 3; ++g)
        fI1[g] = load_frag(ws + WS_WIH1, KPAD, (g * 8 + wave) * 16 + lcol, kb);
    v16h fI2[4][3];
#pragma unroll
    for (int kt = 0; kt < 4; ++kt)
#pragma unroll
        for (int g = 0; g < 3; ++g)
            fI2[kt][g] = load_frag(ws + WS_WIH2, HID, (g * 8 + wave) * 16 + lcol, kt * 32 + kb);

    // per-lane gate biases (depend only on gate column)
    float cbi1[3], cbh1[3], cbi2[3], cbh2[3];
#pragma unroll
    for (int g = 0; g < 3; ++g) {
        int c = g * HID + col;
        cbi1[g] = bi1[c]; cbh1[g] = bh1[c];
        cbi2[g] = bi2[c]; cbh2[g] = bh2[c];
    }

    // previous hidden state, fp32, in WMMA C/D layout (wave-local columns)
    float h1p[2][8] = {}, h2p[2][8] = {};

    __syncthreads();

#pragma unroll 1
    for (int t = 0; t < SEQ; ++t) {
        // ---- stage x_t tile to LDS (f16), prefetch next step ----
        for (int i = tid; i < BT * KIN; i += 256) {
            int b = i / KIN, k = i - b * KIN;
            size_t gofs = ((size_t)(b0 + b) * SEQ + t) * KIN + k;
            sX[b * KPAD + k] = (_Float16)x[gofs];
            if (t + 1 < SEQ) __builtin_prefetch(&x[gofs + KIN], 0, 0);
        }
        __syncthreads();

        // ================= layer 1 =================
        v8f gi[2][3], gh[2][3];
#pragma unroll
        for (int m = 0; m < 2; ++m)
#pragma unroll
            for (int g = 0; g < 3; ++g) { gi[m][g] = bcast8(cbi1[g]); gh[m][g] = bcast8(cbh1[g]); }

        // gi1 = x_t @ w_ih1^T  (K = 32, zero-padded)
#pragma unroll
        for (int m = 0; m < 2; ++m) {
            v16h a = load_frag(sX, KPAD, m * 16 + lcol, kb);
#pragma unroll
            for (int g = 0; g < 3; ++g) gi[m][g] = WMMA(a, fI1[g], gi[m][g]);
        }
        // gh1 = h1 @ w_hh1^T  (K = 128)
#pragma unroll
        for (int kt = 0; kt < 4; ++kt) {
            v16h bw[3];
#pragma unroll
            for (int g = 0; g < 3; ++g)
                bw[g] = load_frag(sW1, HID, (g * 8 + wave) * 16 + lcol, kt * 32 + kb);
#pragma unroll
            for (int m = 0; m < 2; ++m) {
                v16h a = load_frag(sH1, HID, m * 16 + lcol, kt * 32 + kb);
#pragma unroll
                for (int g = 0; g < 3; ++g) gh[m][g] = WMMA(a, bw[g], gh[m][g]);
            }
        }
        // gate combine + relu, write h1 (f16) back for next matmuls
#pragma unroll
        for (int m = 0; m < 2; ++m)
#pragma unroll
            for (int e = 0; e < 8; ++e) {
                float r  = fsigm(gi[m][0][e] + gh[m][0][e]);
                float zz = fsigm(gi[m][1][e] + gh[m][1][e]);
                float nn = ftanh(__builtin_fmaf(r, gh[m][2][e], gi[m][2][e]));
                float h  = fmaxf(__builtin_fmaf(zz, h1p[m][e] - nn, nn), 0.f);
                h1p[m][e] = h;
                sH1[(m * 16 + e + half * 8) * HID + col] = (_Float16)h;
            }
        __syncthreads();

        // ================= layer 2 =================
        v8f g2i[2][3], g2h[2][3];
#pragma unroll
        for (int m = 0; m < 2; ++m)
#pragma unroll
            for (int g = 0; g < 3; ++g) { g2i[m][g] = bcast8(cbi2[g]); g2h[m][g] = bcast8(cbh2[g]); }

#pragma unroll
        for (int kt = 0; kt < 4; ++kt) {
            v16h bw[3];
#pragma unroll
            for (int g = 0; g < 3; ++g)
                bw[g] = load_frag(sW2, HID, (g * 8 + wave) * 16 + lcol, kt * 32 + kb);
#pragma unroll
            for (int m = 0; m < 2; ++m) {
                v16h a1 = load_frag(sH1, HID, m * 16 + lcol, kt * 32 + kb); // gi2 input
                v16h a2 = load_frag(sH2, HID, m * 16 + lcol, kt * 32 + kb); // gh2 input
#pragma unroll
                for (int g = 0; g < 3; ++g) {
                    g2i[m][g] = WMMA(a1, fI2[kt][g], g2i[m][g]);
                    g2h[m][g] = WMMA(a2, bw[g],      g2h[m][g]);
                }
            }
        }
#pragma unroll
        for (int m = 0; m < 2; ++m)
#pragma unroll
            for (int e = 0; e < 8; ++e) {
                float r  = fsigm(g2i[m][0][e] + g2h[m][0][e]);
                float zz = fsigm(g2i[m][1][e] + g2h[m][1][e]);
                float nn = ftanh(__builtin_fmaf(r, g2h[m][2][e], g2i[m][2][e]));
                float h  = fmaxf(__builtin_fmaf(zz, h2p[m][e] - nn, nn), 0.f);
                h2p[m][e] = h;
                sH2[(m * 16 + e + half * 8) * HID + col] = (_Float16)h;
            }
        // top-of-loop barrier covers sH2 write->read hazard across steps
    }
    __syncthreads();

    // ---- output projection: out = h2 @ w_out^T + b_out  (tiny, VALU) ----
    for (int i = tid; i < BT * OUTN; i += 256) {
        int b = i / OUTN, o = i - b * OUTN;
        float s = bout[o];
#pragma unroll 4
        for (int k = 0; k < HID; ++k)
            s += (float)sH2[b * HID + k] * wout[o * HID + k];
        out[(size_t)(b0 + b) * OUTN + o] = s;
    }
}

extern "C" void kernel_launch(void* const* d_in, const int* in_sizes, int n_in,
                              void* d_out, int out_size, void* d_ws, size_t ws_size,
                              hipStream_t stream) {
    const float* x    = (const float*)d_in[0];
    const float* wih1 = (const float*)d_in[1];
    const float* whh1 = (const float*)d_in[2];
    const float* bih1 = (const float*)d_in[3];
    const float* bhh1 = (const float*)d_in[4];
    const float* wih2 = (const float*)d_in[5];
    const float* whh2 = (const float*)d_in[6];
    const float* bih2 = (const float*)d_in[7];
    const float* bhh2 = (const float*)d_in[8];
    const float* wout = (const float*)d_in[9];
    const float* bout = (const float*)d_in[10];
    float* out = (float*)d_out;
    _Float16* ws = (_Float16*)d_ws;

    int n = in_sizes[0] / (SEQ * KIN);   // 8192

    prep_weights<<<(WS_TOTAL + 255) / 256, 256, 0, stream>>>(wih1, whh1, wih2, whh2, ws);

    size_t lds_bytes = (size_t)107520 * sizeof(_Float16); // 215040 B < 320 KB WGP LDS
    gru_main<<<n / BT, 256, lds_bytes, stream>>>(x, ws, bih1, bhh1, bih2, bhh2, wout, bout, out);
}
